// GatedMultiHeadSelfAttention_6038724018437
// MI455X (gfx1250) — compile-verified, matrix operand order checked by reading, not run on hardware
//
#include <hip/hip_runtime.h>
#include <hip/hip_bf16.h>

// ---------------------------------------------------------------------------
// Gated multi-head self-attention, CDNA5 (gfx1250) wave32 WMMA implementation.
// f16 operands / f32 accumulation via V_WMMA_F32_16X16X32_F16.
// ---------------------------------------------------------------------------

#define TB 2
#define TT 2048
#define ED 1024
#define NH 16
#define HD 64
#define BT (TB * TT)   // 4096 rows

typedef __attribute__((ext_vector_type(16))) _Float16 v16h;
typedef __attribute__((ext_vector_type(8)))  float    v8f;

// K-index mapping for 16-bit A fragments (ISA 7.12.2, 16-bit A 16x32):
//   VGPR v (=j/2), element jj (=j%2), lane-half half (=lane/16):
//   K = (v>=4 ? 16 : 0) + half*8 + (v%3..)*2 + jj
// B fragments mirror A with lane = N.
__device__ __forceinline__ int ka_map(int half, int j) {
    int v = j >> 1, jj = j & 1;
    return ((v & 4) << 2) + half * 8 + ((v & 3) << 1) + jj;
}

__device__ __forceinline__ v8f wmma_f16(v16h a, v16h b, v8f c) {
    return __builtin_amdgcn_wmma_f32_16x16x32_f16(
        /*neg_a=*/false, a, /*neg_b=*/false, b,
        /*c_mod=*/(short)0, c, /*reuse_a=*/false, /*reuse_b=*/false);
}

// ---------------------------------------------------------------------------
// Pack hidden states (fp32 [BT,ED]) into A-fragment layout:
//   Xp[rt(256)][kc(32)][lane(32)][16] halves
// ---------------------------------------------------------------------------
__global__ __launch_bounds__(256) void pack_x_kernel(
        const float* __restrict__ hid, _Float16* __restrict__ xp) {
    int rt  = blockIdx.x;          // 16-row tile, 0..255
    int tid = threadIdx.x;
    for (int i = 0; i < 4; ++i) {
        int slot = i * 256 + tid;  // 0..1023 = kc*32 + lane
        int kc = slot >> 5;
        int L  = slot & 31;
        int m = L & 15, half = L >> 4;
        const float* src = hid + (size_t)(rt * 16 + m) * ED + kc * 32;
        v16h frag;
#pragma unroll
        for (int j = 0; j < 16; ++j) frag[j] = (_Float16)src[ka_map(half, j)];
        *(v16h*)(xp + ((size_t)(rt * 32 + kc) * 32 + L) * 16) = frag;
    }
}

// ---------------------------------------------------------------------------
// Pack W_q/W_k/W_v ([H,ED,HD]) into B-fragment layout
//   W*p[h][kc(32)][nt(4)][lane(32)][16]
// and W_o ([H,HD,ED]) into
//   Wop[h][kc(2)][nt(64)][lane(32)][16]
// blockIdx.y selects the matrix (0=q,1=k,2=v,3=o).
// ---------------------------------------------------------------------------
__global__ __launch_bounds__(256) void pack_w_kernel(
        const float* __restrict__ Wq, const float* __restrict__ Wk,
        const float* __restrict__ Wv, const float* __restrict__ Wo,
        _Float16* __restrict__ wqp, _Float16* __restrict__ wkp,
        _Float16* __restrict__ wvp, _Float16* __restrict__ wop) {
    int slot  = blockIdx.x * 256 + threadIdx.x;  // 0..65535
    int which = blockIdx.y;
    int L = slot & 31;
    int n = L & 15, half = L >> 4;
    v16h frag;
    if (which < 3) {
        int nt = (slot >> 5) & 3;
        int kc = (slot >> 7) & 31;
        int h  = slot >> 12;
        const float* W = (which == 0) ? Wq : (which == 1) ? Wk : Wv;
        const float* src = W + (size_t)h * ED * HD;
#pragma unroll
        for (int j = 0; j < 16; ++j) {
            int k = kc * 32 + ka_map(half, j);
            frag[j] = (_Float16)src[(size_t)k * HD + nt * 16 + n];
        }
        _Float16* dst = (which == 0) ? wqp : (which == 1) ? wkp : wvp;
        *(v16h*)(dst + (size_t)slot * 16) = frag;
    } else {
        int nt = (slot >> 5) & 63;
        int kc = (slot >> 11) & 1;
        int h  = slot >> 12;
        const float* src = Wo + (size_t)h * HD * ED;
#pragma unroll
        for (int j = 0; j < 16; ++j) {
            int k = kc * 32 + ka_map(half, j);
            frag[j] = (_Float16)src[(size_t)k * ED + nt * 16 + n];
        }
        *(v16h*)(wop + (size_t)slot * 16) = frag;
    }
}

// ---------------------------------------------------------------------------
// Precompute gated output bias sum_h b_o[h,:]*clip(g[h])/H and zero denomAcc.
// ---------------------------------------------------------------------------
__global__ __launch_bounds__(256) void prep_misc_kernel(
        const float* __restrict__ b_o, const float* __restrict__ gate,
        float* __restrict__ outbias, float* __restrict__ denomAcc) {
    int d = blockIdx.x * 256 + threadIdx.x;  // 0..1023
    float acc = 0.f;
#pragma unroll
    for (int h = 0; h < NH; ++h) {
        float g = fminf(fmaxf(gate[h], 0.f), 1.f);
        acc += b_o[h * ED + d] * g;
    }
    outbias[d] = acc * (1.0f / NH);
    if (d < NH) denomAcc[d] = 0.f;
}

// ---------------------------------------------------------------------------
// QKV projection.  grid = (128 row32-tiles, 16 heads, 3={Q,K,V}), block 256.
// Each wave: one 16x16 output tile (2 m-tiles x 4 e-tiles), K-loop over 1024.
// Results are re-swizzled through LDS into the fragment layouts needed by
// the attention kernel.
// ---------------------------------------------------------------------------
__global__ __launch_bounds__(256) void qkv_kernel(
        const _Float16* __restrict__ xp,
        const _Float16* __restrict__ wqp, const _Float16* __restrict__ wkp,
        const _Float16* __restrict__ wvp,
        const float* __restrict__ bq, const float* __restrict__ bk,
        const float* __restrict__ bv,
        _Float16* __restrict__ Qb, _Float16* __restrict__ Kb,
        _Float16* __restrict__ Vb) {
    __shared__ float tile[32][HD + 1];
    int xb = blockIdx.x;   // 32-row tile over BT: 0..127
    int h  = blockIdx.y;
    int z  = blockIdx.z;   // 0=Q 1=K 2=V
    int tid = threadIdx.x;
    int w = tid >> 5, L = tid & 31;
    int mt = w >> 2, ntile = w & 3;
    const _Float16* wp = (z == 0) ? wqp : (z == 1) ? wkp : wvp;
    const float*  bias = (z == 0) ? bq  : (z == 1) ? bk  : bv;

    int rt16 = xb * 2 + mt;          // global 16-row tile 0..255
    v8f acc = {};
    for (int kc = 0; kc < 32; ++kc) {
        v16h a = *(const v16h*)(xp + ((size_t)(rt16 * 32 + kc) * 32 + L) * 16);
        v16h b = *(const v16h*)(wp + (((size_t)(h * 32 + kc) * 4 + ntile) * 32 + L) * 16);
        acc = wmma_f16(a, b, acc);
    }
    int n = L & 15, half = L >> 4;
    float bb = bias[h * HD + ntile * 16 + n];
#pragma unroll
    for (int r = 0; r < 8; ++r)
        tile[mt * 16 + r + half * 8][ntile * 16 + n] = acc[r] + bb;
    __syncthreads();

    int b_  = xb >> 6;     // batch
    int t32 = xb & 63;     // 32-row tile within batch
    int bh  = b_ * NH + h;
    if (z < 2) {
        // Q as A-fragments / K as B-fragments (same swizzle, lane = m or s).
        if (tid < 128) {
            int ttl = tid >> 6;          // which of the two 16-row tiles
            int kc  = (tid >> 5) & 1;    // e-chunk of 32
            int L2  = tid & 31;
            int mm = L2 & 15, hh = L2 >> 4;
            v16h frag;
#pragma unroll
            for (int j = 0; j < 16; ++j)
                frag[j] = (_Float16)tile[ttl * 16 + mm][kc * 32 + ka_map(hh, j)];
            int ttG = t32 * 2 + ttl;     // 16-row tile within (b,h): 0..127
            _Float16* dst = (z == 0) ? Qb : Kb;
            *(v16h*)(dst + (((size_t)(bh * 128 + ttG) * 2 + kc) * 32 + L2) * 16) = frag;
        }
    } else {
        // V as B-fragments: k = s_local(0..31), n = e.
        if (tid < 128) {
            int nt2 = tid >> 5;          // e-tile 0..3
            int L2  = tid & 31;
            int nn = L2 & 15, hh = L2 >> 4;
            v16h frag;
#pragma unroll
            for (int j = 0; j < 16; ++j) {
                int k = ka_map(hh, j);   // s_local
                frag[j] = (_Float16)tile[k][nt2 * 16 + nn];
            }
            *(v16h*)(Vb + (((size_t)(bh * 64 + t32) * 4 + nt2) * 32 + L2) * 16) = frag;
        }
    }
}

// ---------------------------------------------------------------------------
// Flash attention.  grid = (128 t-tiles, 32 bh), block = 32 (one wave).
// Streaming softmax over 64 chunks of 32 keys; 8 WMMAs per chunk.
// Also accumulates per-head mean L2 row norm and writes the (un-normalized)
// head output as A-fragments for the final projection.
// ---------------------------------------------------------------------------
__global__ __launch_bounds__(32) void attn_kernel(
        const _Float16* __restrict__ Qb, const _Float16* __restrict__ Kb,
        const _Float16* __restrict__ Vb, const float* __restrict__ mask,
        _Float16* __restrict__ Ob, float* __restrict__ denomAcc) {
    __shared__ float lp[16][HD + 1];
    int L  = threadIdx.x;
    int tt = blockIdx.x;   // 16-token tile within (b,h)
    int bh = blockIdx.y;
    int b_ = bh >> 4, h = bh & 15;
    int t0 = tt * 16;
    int n = L & 15, half = L >> 4;
    (void)b_;

    v16h qa0 = *(const v16h*)(Qb + (((size_t)(bh * 128 + tt) * 2 + 0) * 32 + L) * 16);
    v16h qa1 = *(const v16h*)(Qb + (((size_t)(bh * 128 + tt) * 2 + 1) * 32 + L) * 16);

    v8f acc0 = {}, acc1 = {}, acc2 = {}, acc3 = {};
    float mrow[8], lrow[8];
#pragma unroll
    for (int r = 0; r < 8; ++r) { mrow[r] = -3.0e38f; lrow[r] = 0.f; }
    const float scale = 0.125f;  // 1/sqrt(64)

    for (int sc = 0; sc < 64; ++sc) {
        v8f S[2];
#pragma unroll
        for (int t2 = 0; t2 < 2; ++t2) {
            int st = sc * 2 + t2;
            v16h kb0 = *(const v16h*)(Kb + (((size_t)(bh * 128 + st) * 2 + 0) * 32 + L) * 16);
            v16h kb1 = *(const v16h*)(Kb + (((size_t)(bh * 128 + st) * 2 + 1) * 32 + L) * 16);
            v8f s = {};
            s = wmma_f16(qa0, kb0, s);
            s = wmma_f16(qa1, kb1, s);
#pragma unroll
            for (int r = 0; r < 8; ++r) {
                int m = r + half * 8;
                s[r] = s[r] * scale + mask[(size_t)(t0 + m) * TT + st * 16 + n];
            }
            S[t2] = s;
        }
        // streaming softmax update (rows live across the 16-lane halves)
#pragma unroll
        for (int r = 0; r < 8; ++r) {
            float mx = fmaxf(S[0][r], S[1][r]);
            mx = fmaxf(mx, __shfl_xor(mx, 1, 32));
            mx = fmaxf(mx, __shfl_xor(mx, 2, 32));
            mx = fmaxf(mx, __shfl_xor(mx, 4, 32));
            mx = fmaxf(mx, __shfl_xor(mx, 8, 32));
            float mnew  = fmaxf(mrow[r], mx);
            float alpha = __expf(mrow[r] - mnew);
            mrow[r] = mnew;
            S[0][r] = __expf(S[0][r] - mnew);
            S[1][r] = __expf(S[1][r] - mnew);
            float rs = S[0][r] + S[1][r];
            rs += __shfl_xor(rs, 1, 32);
            rs += __shfl_xor(rs, 2, 32);
            rs += __shfl_xor(rs, 4, 32);
            rs += __shfl_xor(rs, 8, 32);
            lrow[r] = lrow[r] * alpha + rs;
            acc0[r] *= alpha; acc1[r] *= alpha; acc2[r] *= alpha; acc3[r] *= alpha;
        }
        // D-layout P -> LDS -> A-fragment (f16)
#pragma unroll
        for (int r = 0; r < 8; ++r) {
            lp[r + half * 8][n]      = S[0][r];
            lp[r + half * 8][16 + n] = S[1][r];
        }
        __syncthreads();
        v16h pa;
#pragma unroll
        for (int j = 0; j < 16; ++j)
            pa[j] = (_Float16)lp[L & 15][ka_map(half, j)];
        v16h vb0 = *(const v16h*)(Vb + (((size_t)(bh * 64 + sc) * 4 + 0) * 32 + L) * 16);
        v16h vb1 = *(const v16h*)(Vb + (((size_t)(bh * 64 + sc) * 4 + 1) * 32 + L) * 16);
        v16h vb2 = *(const v16h*)(Vb + (((size_t)(bh * 64 + sc) * 4 + 2) * 32 + L) * 16);
        v16h vb3 = *(const v16h*)(Vb + (((size_t)(bh * 64 + sc) * 4 + 3) * 32 + L) * 16);
        acc0 = wmma_f16(pa, vb0, acc0);
        acc1 = wmma_f16(pa, vb1, acc1);
        acc2 = wmma_f16(pa, vb2, acc2);
        acc3 = wmma_f16(pa, vb3, acc3);
        __syncthreads();
    }

    // finalize: divide by l, stage 16x64 out tile in LDS, row L2 norms
    float ss[8];
#pragma unroll
    for (int r = 0; r < 8; ++r) {
        float li = 1.0f / lrow[r];
        float o0 = acc0[r] * li, o1 = acc1[r] * li;
        float o2 = acc2[r] * li, o3 = acc3[r] * li;
        int m = r + half * 8;
        lp[m][n] = o0; lp[m][16 + n] = o1; lp[m][32 + n] = o2; lp[m][48 + n] = o3;
        ss[r] = o0 * o0 + o1 * o1 + o2 * o2 + o3 * o3;
    }
    __syncthreads();
    float tot = 0.f;
#pragma unroll
    for (int r = 0; r < 8; ++r) {
        float s2 = ss[r];
        s2 += __shfl_xor(s2, 1, 32);
        s2 += __shfl_xor(s2, 2, 32);
        s2 += __shfl_xor(s2, 4, 32);
        s2 += __shfl_xor(s2, 8, 32);
        tot += sqrtf(s2);
    }
    tot += __shfl_xor(tot, 16, 32);            // combine both halves (16 rows)
    if (L == 0) atomicAdd(denomAcc + h, tot * (1.0f / BT));

    // write un-normalized out tile as A-fragments for the projection GEMM
    int rtG = (bh >> 4) * 128 + tt;            // global 16-row tile 0..255
#pragma unroll
    for (int kc = 0; kc < 2; ++kc) {
        v16h o16;
#pragma unroll
        for (int j = 0; j < 16; ++j)
            o16[j] = (_Float16)lp[L & 15][kc * 32 + ka_map(half, j)];
        *(v16h*)(Ob + (((size_t)(rtG * 16 + h) * 2 + kc) * 32 + L) * 16) = o16;
    }
}

// ---------------------------------------------------------------------------
// Output projection: one K=1024 GEMM over concatenated heads, with per-head
// clip(gate)/max(denom,1e-5)/H folded into the A fragments.
// grid = (256 row tiles, 16 d-groups of 64), block 128 (4 waves).
// ---------------------------------------------------------------------------
__global__ __launch_bounds__(128) void proj_kernel(
        const _Float16* __restrict__ Ob, const _Float16* __restrict__ wop,
        const float* __restrict__ outbias, const float* __restrict__ denomAcc,
        const float* __restrict__ gate, float* __restrict__ out) {
    int rt = blockIdx.x, dg = blockIdx.y;
    int tid = threadIdx.x;
    int w = tid >> 5, L = tid & 31;
    int ntg = dg * 4 + w;            // 16-col tile 0..63
    int n = L & 15, half = L >> 4;
    v8f acc = {};
    for (int c = 0; c < 32; ++c) {
        int h = c >> 1, kc = c & 1;
        float g  = fminf(fmaxf(gate[h], 0.f), 1.f);
        float dn = fmaxf(denomAcc[h], 1e-5f);
        float s  = g / (dn * (float)NH);
        v16h a = *(const v16h*)(Ob + (((size_t)(rt * 16 + h) * 2 + kc) * 32 + L) * 16);
#pragma unroll
        for (int j = 0; j < 16; ++j) a[j] = (_Float16)((float)a[j] * s);
        v16h bf = *(const v16h*)(wop + (((size_t)(h * 2 + kc) * 64 + ntg) * 32 + L) * 16);
        acc = wmma_f16(a, bf, acc);
    }
    int d = ntg * 16 + n;
    float bsum = outbias[d];
#pragma unroll
    for (int r = 0; r < 8; ++r) {
        int row = rt * 16 + r + half * 8;
        out[(size_t)row * ED + d] = acc[r] + bsum;
    }
}

// ---------------------------------------------------------------------------
// Host launcher.
// ---------------------------------------------------------------------------
extern "C" void kernel_launch(void* const* d_in, const int* in_sizes, int n_in,
                              void* d_out, int out_size, void* d_ws, size_t ws_size,
                              hipStream_t stream) {
    const float* hid  = (const float*)d_in[0];
    const float* mask = (const float*)d_in[1];
    const float* Wq   = (const float*)d_in[2];
    const float* bq   = (const float*)d_in[3];
    const float* Wk   = (const float*)d_in[4];
    const float* bk   = (const float*)d_in[5];
    const float* Wv   = (const float*)d_in[6];
    const float* bv   = (const float*)d_in[7];
    const float* Wo   = (const float*)d_in[8];
    const float* bo   = (const float*)d_in[9];
    const float* gate = (const float*)d_in[10];
    float* out = (float*)d_out;

    char* ws = (char*)d_ws;
    const size_t MB = 1u << 20;
    _Float16* Xp  = (_Float16*)(ws);             // 8 MB  packed hidden (A frags)
    _Float16* Wqp = (_Float16*)(ws +  8 * MB);   // 2 MB
    _Float16* Wkp = (_Float16*)(ws + 10 * MB);   // 2 MB
    _Float16* Wvp = (_Float16*)(ws + 12 * MB);   // 2 MB
    _Float16* Wop = (_Float16*)(ws + 14 * MB);   // 2 MB
    _Float16* Qb  = (_Float16*)(ws + 16 * MB);   // 8 MB
    _Float16* Kb  = (_Float16*)(ws + 24 * MB);   // 8 MB
    _Float16* Vb  = (_Float16*)(ws + 32 * MB);   // 8 MB
    float* outbias  = (float*)(ws + 40 * MB);    // 4 KB
    float* denomAcc = (float*)(ws + 40 * MB + 4096);
    _Float16* Ob = Xp;  // alias: Xp dead after qkv_kernel, Ob written by attn

    pack_x_kernel<<<dim3(256), dim3(256), 0, stream>>>(hid, Xp);
    pack_w_kernel<<<dim3(256, 4), dim3(256), 0, stream>>>(Wq, Wk, Wv, Wo,
                                                          Wqp, Wkp, Wvp, Wop);
    prep_misc_kernel<<<dim3(4), dim3(256), 0, stream>>>(bo, gate, outbias, denomAcc);
    qkv_kernel<<<dim3(128, 16, 3), dim3(256), 0, stream>>>(Xp, Wqp, Wkp, Wvp,
                                                           bq, bk, bv, Qb, Kb, Vb);
    attn_kernel<<<dim3(128, 32), dim3(32), 0, stream>>>(Qb, Kb, Vb, mask, Ob, denomAcc);
    proj_kernel<<<dim3(256, 16), dim3(128), 0, stream>>>(Ob, Wop, outbias, denomAcc,
                                                         gate, out);
    (void)in_sizes; (void)n_in; (void)out_size; (void)ws_size;
}
